// SparseUNet_661424964111
// MI455X (gfx1250) — matrix operand order repaired and checked by reading, not compiled, not run on hardware
//
#include <hip/hip_runtime.h>

// ---------------------------------------------------------------------------
// CDNA5 (gfx1250) 3D U-Net forward pass.
// Conv3x3x3: tap-major implicit GEMM on v_wmma_f32_16x16x32_f16 with an
// LDS-staged halo tile (stage once, reuse for the whole K loop and all
// N-tiles). f16 inputs, f32 accumulate. wave32 throughout.
// ---------------------------------------------------------------------------

typedef __attribute__((ext_vector_type(16))) _Float16 v16h;
typedef __attribute__((ext_vector_type(8)))  _Float16 v8h;
typedef __attribute__((ext_vector_type(8)))  float    v8f;

#define BN_EPS 1e-5f

// ---------------------------------------------------------------------------
// Weight pre-pack: w[co][ci][tap] (f32, OIDHW) -> w16[co16][K32] (f16),
// K-order k = tap*Cin + ci, zero padded in both co and k.
// grid = (ceil(K32/256), co16)
// ---------------------------------------------------------------------------
__global__ void pack_w16(const float* __restrict__ w, _Float16* __restrict__ w16,
                         int Cin, int cinShift, int Cout, int K32)
{
    const int k  = (int)blockIdx.x * 256 + (int)threadIdx.x;
    const int co = (int)blockIdx.y;
    if (k >= K32) return;
    const int tap = k >> cinShift;
    const int ci  = k & (Cin - 1);
    float v = 0.f;
    if (co < Cout && tap < 27)
        v = w[((size_t)co * Cin + ci) * 27 + tap];
    w16[(size_t)co * K32 + k] = (_Float16)v;
}

// ---------------------------------------------------------------------------
// Tiled implicit-GEMM conv3x3x3 "same".
//   Workgroup: 256 threads = 8 waves, one spatial tile (TDxTHx16 outputs).
//   Wave w: M-subtile = w % MSUB (one 16-voxel x-run), N-group = w / MSUB.
//   A fragment (16-bit A 16x32, ISA 7.12.2): two 8-half K-runs per lane,
//   each run = single tap, 8 consecutive ci -> one ds_load_b128 from the
//   staged halo tile. B fragment: two 8-half k-runs from packed w16, same k as A.
//   D writeback: each lane owns 8 consecutive-x f32 at fixed co -> one
//   32-byte store when the tile is fully inside W.
// ---------------------------------------------------------------------------
template <int CIN, int COUT>
__global__ __launch_bounds__(256) void conv3_tile(
    const float* __restrict__ in, const _Float16* __restrict__ w16,
    const float* __restrict__ bias, float* __restrict__ out,
    int N, int D, int H, int W, int hasBias)
{
    constexpr int TD  = (CIN >= 128) ? 1 : 2;
    constexpr int TH  = (CIN >= 128) ? 2 : 4;
    constexpr int TW  = 16;
    constexpr int MSUB = TD * TH;          // 16-voxel M-subtiles per tile
    constexpr int NWG  = 8 / MSUB;         // wave groups over N tiles
    constexpr int HD = TD + 2, HH = TH + 2, HX = TW + 2;
    constexpr int HV = HD * HH * HX;       // halo voxels
    constexpr int K   = 27 * CIN;
    constexpr int K32 = (K + 31) & ~31;
    constexpr int NT  = (COUT + 15) / 16;
    constexpr int NTL = (NT + NWG - 1) / NWG;
    constexpr int CSH = (CIN >= 128) ? 7 : (CIN >= 64) ? 6 : (CIN >= 32) ? 5
                      : (CIN >= 16) ? 4 : (CIN >= 8) ? 3 : 0;

    __shared__ _Float16 smem[HV * CIN];    // [vox][ci], ci contiguous

    const int HWg = H * W, DHWg = D * HWg;
    // block -> (n, z0, y0, x0)
    const int tx = (W + TW - 1) / TW, ty = H / TH, tz = D / TD;
    int b = (int)blockIdx.x;
    const int tix = b % tx; b /= tx;
    const int tiy = b % ty; b /= ty;
    const int tiz = b % tz; const int n = b / tz;
    const int x0 = tix * TW, y0 = tiy * TH, z0 = tiz * TD;

    // ---- stage halo tile (f32 -> f16), coalesced along x ----
    for (int i = (int)threadIdx.x; i < HV * CIN; i += 256) {
        const int ci  = i / HV;
        const int vox = i - ci * HV;
        int v = vox;
        const int hx = v % HX; v /= HX;
        const int hy = v % HH; const int hz = v / HH;
        const int gz = z0 + hz - 1, gy = y0 + hy - 1, gx = x0 + hx - 1;
        float val = 0.f;
        if ((unsigned)gz < (unsigned)D && (unsigned)gy < (unsigned)H &&
            (unsigned)gx < (unsigned)W)
            val = in[(size_t)(n * CIN + ci) * DHWg + gz * HWg + gy * W + gx];
        smem[vox * CIN + ci] = (_Float16)val;
    }
    __syncthreads();

    const int wid  = (int)threadIdx.x >> 5;
    const int lane = (int)threadIdx.x & 31;
    const int hi = lane >> 4, lo = lane & 15;
    const int msub = wid % MSUB;
    const int grp  = wid / MSUB;
    const int tzl = msub / TH, tyl = msub % TH;
    // lane's voxel (row r = x = lo) at halo coords (tzl+1, tyl+1, lo+1);
    // tap (kd,kh,kw) address = base0 + (kd*HH + kh)*HX + kw
    const int base0 = (tzl * HH + tyl) * HX + lo;

    v8h zf;
#pragma unroll
    for (int t = 0; t < 8; ++t) zf[t] = (_Float16)0.f;

    v8f acc[NTL];
#pragma unroll
    for (int j = 0; j < NTL; ++j) acc[j] = {0.f,0.f,0.f,0.f,0.f,0.f,0.f,0.f};

    for (int kc = 0; kc < K32; kc += 32) {
        v16h a;
        if (CIN >= 8) {
            // run 0: elements 0..7, k = kc + 8*hi + [0,8)
            // run 1: elements 8..15, k = kc + 16 + 8*hi + [0,8)
            const int kb0 = kc + 8 * hi;
            const int kb1 = kb0 + 16;
            const int tap0 = kb0 >> CSH, ci0 = kb0 & (CIN - 1);
            const int tap1 = kb1 >> CSH, ci1 = kb1 & (CIN - 1);
            int td0 = tap0 / 9, r0 = tap0 - td0 * 9;
            int th0 = r0 / 3,  tw0 = r0 - th0 * 3;
            int td1 = tap1 / 9, r1 = tap1 - td1 * 9;
            int th1 = r1 / 3,  tw1 = r1 - th1 * 3;
            v8h a0 = zf, a1 = zf;
            if (tap0 < 27)
                a0 = *(const v8h*)&smem[(base0 + (td0 * HH + th0) * HX + tw0) * CIN + ci0];
            if (tap1 < 27)
                a1 = *(const v8h*)&smem[(base0 + (td1 * HH + th1) * HX + tw1) * CIN + ci1];
            a = __builtin_shufflevector(a0, a1, 0, 1, 2, 3, 4, 5, 6, 7,
                                        8, 9, 10, 11, 12, 13, 14, 15);
        } else {
            // CIN == 1: k == tap, per-element gather (first layer only)
#pragma unroll
            for (int e = 0; e < 16; ++e) {
                const int j2 = e >> 1, hh2 = e & 1;
                const int k = kc + 2 * j2 + hh2 + 8 * ((j2 >> 2) + hi);
                _Float16 v = (_Float16)0.f;
                if (k < 27) {
                    const int kd = k / 9, rr = k - kd * 9;
                    const int kh = rr / 3, kw = rr - kh * 3;
                    v = smem[base0 + (kd * HH + kh) * HX + kw];
                }
                a[e] = v;
            }
        }
#pragma unroll
        for (int j = 0; j < NTL; ++j) {
            const int nt = grp + j * NWG;       // wave-uniform
            if (nt < NT) {
                const int co = nt * 16 + lo;
                const v8h b0 = *(const v8h*)&w16[(size_t)co * K32 + kc + 8 * hi];
                const v8h b1 = *(const v8h*)&w16[(size_t)co * K32 + kc + 16 + 8 * hi];
                const v16h bf = __builtin_shufflevector(b0, b1, 0, 1, 2, 3, 4, 5, 6, 7,
                                                        8, 9, 10, 11, 12, 13, 14, 15);
                acc[j] = __builtin_amdgcn_wmma_f32_16x16x32_f16(
                             false, a, false, bf, (short)0, acc[j], false, false);
            }
        }
    }

    // ---- store (C/D layout: VGPR v -> M = v + 8*hi = x, N = lo = co) ----
    const int gz = z0 + tzl, gy = y0 + tyl;
    const size_t rowBase = (size_t)n * COUT * DHWg + (size_t)gz * HWg + (size_t)gy * W;
#pragma unroll
    for (int j = 0; j < NTL; ++j) {
        const int nt = grp + j * NWG;
        if (nt >= NT) continue;
        const int co = nt * 16 + lo;
        if (co >= COUT) continue;
        const float bv = hasBias ? bias[co] : 0.f;
        v8f r = acc[j];
#pragma unroll
        for (int v = 0; v < 8; ++v) r[v] += bv;
        float* p = out + rowBase + (size_t)co * DHWg + x0 + 8 * hi;
        if (x0 + TW <= W) {
            *(v8f*)p = r;                         // 32B aligned -> 2x b128
        } else {
#pragma unroll
            for (int v = 0; v < 8; ++v) {
                const int x = x0 + v + 8 * hi;
                if (x < W) p[v] = r[v];
            }
        }
    }
}

// ---------------------------------------------------------------------------
// Transposed conv, kernel 2 / stride 2 / lhs_dilation 2 / pad 1:
// out[o] = sum_ci x[ci, o>>1] * w[co, ci, 1-(o&1), ...]
// ---------------------------------------------------------------------------
__global__ void upconv2x(const float* __restrict__ in, const float* __restrict__ w,
                         float* __restrict__ out, int N, int Cin, int Cout,
                         int Do, int Ho, int Wo)
{
    const long total = (long)N * Cout * Do * Ho * Wo;
    long idx = (long)blockIdx.x * blockDim.x + threadIdx.x;
    if (idx >= total) return;
    int ow = (int)(idx % Wo); long t = idx / Wo;
    int oh = (int)(t % Ho); t /= Ho;
    int od = (int)(t % Do); t /= Do;
    int co = (int)(t % Cout); int n = (int)(t / Cout);
    const int Di = Do >> 1, Hi = Ho >> 1, Wi = Wo >> 1;
    const int id = od >> 1, ih = oh >> 1, iw = ow >> 1;
    const int kd = 1 - (od & 1), kh = 1 - (oh & 1), kw = 1 - (ow & 1);
    const long chs = (long)Di * Hi * Wi;
    const float* ip = in + (size_t)n * Cin * chs + (long)id * Hi * Wi + ih * Wi + iw;
    const float* wp = w + (size_t)co * Cin * 8 + kd * 4 + kh * 2 + kw;
    float acc = 0.f;
    for (int ci = 0; ci < Cin; ++ci)
        acc += ip[(size_t)ci * chs] * wp[(size_t)ci * 8];
    out[idx] = acc;
}

// ---------------------------------------------------------------------------
// 2x2x2 max pool, stride 2
// ---------------------------------------------------------------------------
__global__ void maxpool2x(const float* __restrict__ in, float* __restrict__ out,
                          int NC, int Di, int Hi, int Wi)
{
    const int Do = Di >> 1, Ho = Hi >> 1, Wo = Wi >> 1;
    const long total = (long)NC * Do * Ho * Wo;
    long idx = (long)blockIdx.x * blockDim.x + threadIdx.x;
    if (idx >= total) return;
    int ow = (int)(idx % Wo); long t = idx / Wo;
    int oh = (int)(t % Ho); t /= Ho;
    int od = (int)(t % Do); int nc = (int)(t / Do);
    const long z = (long)Hi * Wi;
    const long base = ((size_t)nc * Di + 2 * od) * z + (2 * oh) * Wi + 2 * ow;
    float m = in[base];
    m = fmaxf(m, in[base + 1]);
    m = fmaxf(m, in[base + Wi]);
    m = fmaxf(m, in[base + Wi + 1]);
    m = fmaxf(m, in[base + z]);
    m = fmaxf(m, in[base + z + 1]);
    m = fmaxf(m, in[base + z + Wi]);
    m = fmaxf(m, in[base + z + Wi + 1]);
    out[idx] = m;
}

// ---------------------------------------------------------------------------
// Training-mode BatchNorm, deterministic two-stage reduction (no FP atomics).
// ---------------------------------------------------------------------------
__global__ __launch_bounds__(256) void bn_stats_partial(
    const float* __restrict__ x, float* __restrict__ part,
    int N, int C, long DHW, int chunks)
{
    __shared__ float ssum[256];
    __shared__ float ssq[256];
    const int c     = (int)blockIdx.x % C;
    const int chunk = (int)blockIdx.x / C;
    const long stride = (long)chunks * blockDim.x;
    float s = 0.f, q = 0.f;
    for (long r = (long)chunk * blockDim.x + threadIdx.x; r < DHW; r += stride) {
        for (int nn = 0; nn < N; ++nn) {
            const float v = x[((size_t)nn * C + c) * DHW + r];
            s += v; q += v * v;
        }
    }
    ssum[threadIdx.x] = s; ssq[threadIdx.x] = q;
    __syncthreads();
    for (int off = 128; off > 0; off >>= 1) {
        if ((int)threadIdx.x < off) {
            ssum[threadIdx.x] += ssum[threadIdx.x + off];
            ssq[threadIdx.x]  += ssq[threadIdx.x + off];
        }
        __syncthreads();
    }
    if (threadIdx.x == 0) {
        part[2 * ((size_t)c * chunks + chunk) + 0] = ssum[0];
        part[2 * ((size_t)c * chunks + chunk) + 1] = ssq[0];
    }
}

__global__ void bn_reduce(const float* __restrict__ part, float* __restrict__ stats,
                          int C, int chunks)
{
    int c = (int)blockIdx.x * blockDim.x + threadIdx.x;
    if (c >= C) return;
    float s = 0.f, q = 0.f;
    for (int i = 0; i < chunks; ++i) {
        s += part[2 * ((size_t)c * chunks + i) + 0];
        q += part[2 * ((size_t)c * chunks + i) + 1];
    }
    stats[2 * c + 0] = s;
    stats[2 * c + 1] = q;
}

__global__ void bn_apply(float* __restrict__ x, const float* __restrict__ stats,
                         const float* __restrict__ g, const float* __restrict__ b,
                         int C, long DHW, long total, float invCnt, int relu)
{
    long idx = (long)blockIdx.x * blockDim.x + threadIdx.x;
    if (idx >= total) return;
    const int c = (int)((idx / DHW) % C);
    const float mean = stats[2 * c] * invCnt;
    const float var  = stats[2 * c + 1] * invCnt - mean * mean;
    float y = (x[idx] - mean) * rsqrtf(var + BN_EPS) * g[c] + b[c];
    if (relu) y = fmaxf(y, 0.f);
    x[idx] = y;
}

// Channel-block copy for concatenation (NCDHW).
__global__ void copy_ch(float* __restrict__ dst, const float* __restrict__ src,
                        int N, int Cdst, int coff, int Csrc, long DHW)
{
    const long total = (long)N * Csrc * DHW;
    long idx = (long)blockIdx.x * blockDim.x + threadIdx.x;
    if (idx >= total) return;
    const long i = idx % DHW; long t = idx / DHW;
    const int c = (int)(t % Csrc); const int n = (int)(t / Csrc);
    dst[((size_t)n * Cdst + coff + c) * DHW + i] = src[idx];
}

// ---------------------------------------------------------------------------
// Host-side orchestration
// ---------------------------------------------------------------------------
static inline int cdiv(long a, long b) { return (int)((a + b - 1) / b); }

template <int CIN, int COUT>
static void launch_conv(const float* in, const _Float16* w16, const float* bias,
                        float* out, int N, int D, int H, int W, int hasBias,
                        hipStream_t s)
{
    constexpr int TD = (CIN >= 128) ? 1 : 2;
    constexpr int TH = (CIN >= 128) ? 2 : 4;
    const int blocks = N * (D / TD) * (H / TH) * ((W + 15) / 16);
    conv3_tile<CIN, COUT><<<blocks, 256, 0, s>>>(in, w16, bias, out, N, D, H, W,
                                                 hasBias);
}

static void run_conv(const float* in, const float* w, const float* bias, float* out,
                     int N, int Cin, int Cout, int D, int H, int W, int hasBias,
                     _Float16* w16, hipStream_t s)
{
    const int K32 = ((27 * Cin) + 31) & ~31;
    int csh = 0; while ((1 << csh) < Cin) ++csh;
    const int co16 = ((Cout + 15) / 16) * 16;
    dim3 g((unsigned)cdiv(K32, 256), (unsigned)co16);
    pack_w16<<<g, 256, 0, s>>>(w, w16, Cin, csh, Cout, K32);
#define CONV_CASE(ci, co)                                                       \
    if (Cin == ci && Cout == co) {                                              \
        launch_conv<ci, co>(in, w16, bias, out, N, D, H, W, hasBias, s);        \
        return;                                                                 \
    }
    CONV_CASE(1, 8)   CONV_CASE(8, 8)    CONV_CASE(8, 16)  CONV_CASE(16, 16)
    CONV_CASE(16, 32) CONV_CASE(32, 32)  CONV_CASE(32, 64) CONV_CASE(64, 64)
    CONV_CASE(64, 128) CONV_CASE(128, 128) CONV_CASE(128, 64) CONV_CASE(64, 32)
    CONV_CASE(32, 16) CONV_CASE(16, 8)   CONV_CASE(8, 32)
#undef CONV_CASE
}

static void run_bn(float* x, int N, int C, int D, int H, int W,
                   const float* g, const float* b, int relu,
                   float* stats, float* part, hipStream_t s)
{
    const long DHW = (long)D * H * W;
    const long per = (long)N * DHW;
    int chunks = (int)(DHW / 16384); if (chunks < 1) chunks = 1; if (chunks > 128) chunks = 128;
    bn_stats_partial<<<C * chunks, 256, 0, s>>>(x, part, N, C, DHW, chunks);
    bn_reduce<<<cdiv(C, 128), 128, 0, s>>>(part, stats, C, chunks);
    const long total = (long)C * per;
    bn_apply<<<cdiv(total, 256), 256, 0, s>>>(x, stats, g, b, C, DHW, total,
                                              1.0f / (float)per, relu);
}

static void run_pool(const float* in, float* out, int NC, int D, int H, int W,
                     hipStream_t s)
{
    const long total = (long)NC * (D >> 1) * (H >> 1) * (W >> 1);
    maxpool2x<<<cdiv(total, 256), 256, 0, s>>>(in, out, NC, D, H, W);
}

static void run_upconv(const float* in, const float* w, float* out,
                       int N, int Cin, int Cout, int Do, int Ho, int Wo,
                       hipStream_t s)
{
    const long total = (long)N * Cout * Do * Ho * Wo;
    upconv2x<<<cdiv(total, 256), 256, 0, s>>>(in, w, out, N, Cin, Cout, Do, Ho, Wo);
}

static void run_copy(float* dst, const float* src, int N, int Cdst, int coff,
                     int Csrc, long DHW, hipStream_t s)
{
    const long total = (long)N * Csrc * DHW;
    copy_ch<<<cdiv(total, 256), 256, 0, s>>>(dst, src, N, Cdst, coff, Csrc, DHW);
}

extern "C" void kernel_launch(void* const* d_in, const int* in_sizes, int n_in,
                              void* d_out, int out_size, void* d_ws, size_t ws_size,
                              hipStream_t stream)
{
    (void)in_sizes; (void)n_in; (void)out_size; (void)ws_size;
    const float* X = (const float*)d_in[0];
    #define F(i) ((const float*)d_in[(i)])

    // -------- workspace layout (bump allocated, 256B aligned) --------
    char*  ws  = (char*)d_ws;
    size_t off = 0;
    auto alloc = [&](size_t bytes) -> char* {
        char* p = ws + off;
        off += (bytes + 255) & ~(size_t)255;
        return p;
    };
    const int    N  = 4;
    const size_t L0 = 64 * 64 * 64, L1 = 32 * 32 * 32, L2 = 16 * 16 * 16,
                 L3 = 8 * 8 * 8;
    float*     OG1   = (float*)alloc((size_t)N * 8  * L0 * 4);   // 32 MiB skip
    float*     OG2   = (float*)alloc((size_t)N * 16 * L1 * 4);   // 8 MiB skip
    float*     OG3   = (float*)alloc((size_t)N * 32 * L2 * 4);   // 2 MiB skip
    float*     OG4   = (float*)alloc((size_t)N * 64 * L3 * 4);   // 0.5 MiB skip
    float*     STATS = (float*)alloc(2 * 128 * sizeof(float));
    float*     PART  = (float*)alloc(2 * 128 * 128 * sizeof(float));
    _Float16*  W16   = (_Float16*)alloc(1 << 20);                // packed weights
    float*     A     = (float*)alloc((size_t)N * 8  * L0 * 4);   // 32 MiB ping
    float*     B     = (float*)alloc((size_t)N * 8  * L0 * 4);   // 32 MiB pong
    float*     CC    = (float*)alloc((size_t)N * 16 * L0 * 4);   // 64 MiB concat

    // ---------------- down 1 (64^3) ----------------
    run_conv(X, F(1), nullptr, A, N, 1, 8, 64, 64, 64, 0, W16, stream);
    run_bn(A, N, 8, 64, 64, 64, F(2), F(3), 1, STATS, PART, stream);
    run_conv(A, F(4), nullptr, OG1, N, 8, 8, 64, 64, 64, 0, W16, stream);
    run_bn(OG1, N, 8, 64, 64, 64, F(5), F(6), 0, STATS, PART, stream);
    run_pool(OG1, A, N * 8, 64, 64, 64, stream);

    // ---------------- down 2 (32^3) ----------------
    run_conv(A, F(7), nullptr, B, N, 8, 16, 32, 32, 32, 0, W16, stream);
    run_bn(B, N, 16, 32, 32, 32, F(8), F(9), 1, STATS, PART, stream);
    run_conv(B, F(10), nullptr, OG2, N, 16, 16, 32, 32, 32, 0, W16, stream);
    run_bn(OG2, N, 16, 32, 32, 32, F(11), F(12), 0, STATS, PART, stream);
    run_pool(OG2, A, N * 16, 32, 32, 32, stream);

    // ---------------- down 3 (16^3) ----------------
    run_conv(A, F(13), nullptr, B, N, 16, 32, 16, 16, 16, 0, W16, stream);
    run_bn(B, N, 32, 16, 16, 16, F(14), F(15), 1, STATS, PART, stream);
    run_conv(B, F(16), nullptr, OG3, N, 32, 32, 16, 16, 16, 0, W16, stream);
    run_bn(OG3, N, 32, 16, 16, 16, F(17), F(18), 0, STATS, PART, stream);
    run_pool(OG3, A, N * 32, 16, 16, 16, stream);

    // ---------------- down 4 (8^3) ----------------
    run_conv(A, F(19), nullptr, B, N, 32, 64, 8, 8, 8, 0, W16, stream);
    run_bn(B, N, 64, 8, 8, 8, F(20), F(21), 1, STATS, PART, stream);
    run_conv(B, F(22), nullptr, OG4, N, 64, 64, 8, 8, 8, 0, W16, stream);
    run_bn(OG4, N, 64, 8, 8, 8, F(23), F(24), 0, STATS, PART, stream);
    run_pool(OG4, A, N * 64, 8, 8, 8, stream);

    // ---------------- bottleneck (4^3) ----------------
    run_conv(A, F(25), nullptr, B, N, 64, 128, 4, 4, 4, 0, W16, stream);
    run_bn(B, N, 128, 4, 4, 4, F(26), F(27), 0, STATS, PART, stream);
    run_conv(B, F(28), nullptr, A, N, 128, 128, 4, 4, 4, 0, W16, stream);
    run_bn(A, N, 128, 4, 4, 4, F(29), F(30), 0, STATS, PART, stream);

    // ---------------- up 4 (-> 8^3) ----------------
    run_upconv(A, F(31), B, N, 128, 64, 8, 8, 8, stream);
    run_bn(B, N, 64, 8, 8, 8, F(32), F(33), 1, STATS, PART, stream);
    run_copy(CC, B,   N, 128, 0,  64, L3, stream);
    run_copy(CC, OG4, N, 128, 64, 64, L3, stream);
    run_conv(CC, F(34), nullptr, A, N, 128, 64, 8, 8, 8, 0, W16, stream);
    run_bn(A, N, 64, 8, 8, 8, F(35), F(36), 1, STATS, PART, stream);
    run_conv(A, F(37), nullptr, B, N, 64, 64, 8, 8, 8, 0, W16, stream);
    run_bn(B, N, 64, 8, 8, 8, F(38), F(39), 1, STATS, PART, stream);

    // ---------------- up 3 (-> 16^3) ----------------
    run_upconv(B, F(40), A, N, 64, 32, 16, 16, 16, stream);
    run_bn(A, N, 32, 16, 16, 16, F(41), F(42), 1, STATS, PART, stream);
    run_copy(CC, A,   N, 64, 0,  32, L2, stream);
    run_copy(CC, OG3, N, 64, 32, 32, L2, stream);
    run_conv(CC, F(43), nullptr, B, N, 64, 32, 16, 16, 16, 0, W16, stream);
    run_bn(B, N, 32, 16, 16, 16, F(44), F(45), 1, STATS, PART, stream);
    run_conv(B, F(46), nullptr, A, N, 32, 32, 16, 16, 16, 0, W16, stream);
    run_bn(A, N, 32, 16, 16, 16, F(47), F(48), 1, STATS, PART, stream);

    // ---------------- up 2 (-> 32^3) ----------------
    run_upconv(A, F(49), B, N, 32, 16, 32, 32, 32, stream);
    run_bn(B, N, 16, 32, 32, 32, F(50), F(51), 1, STATS, PART, stream);
    run_copy(CC, B,   N, 32, 0,  16, L1, stream);
    run_copy(CC, OG2, N, 32, 16, 16, L1, stream);
    run_conv(CC, F(52), nullptr, A, N, 32, 16, 32, 32, 32, 0, W16, stream);
    run_bn(A, N, 16, 32, 32, 32, F(53), F(54), 1, STATS, PART, stream);
    run_conv(A, F(55), nullptr, B, N, 16, 16, 32, 32, 32, 0, W16, stream);
    run_bn(B, N, 16, 32, 32, 32, F(56), F(57), 1, STATS, PART, stream);

    // ---------------- up 1 (-> 64^3) ----------------
    run_upconv(B, F(58), A, N, 16, 8, 64, 64, 64, stream);
    run_bn(A, N, 8, 64, 64, 64, F(59), F(60), 1, STATS, PART, stream);
    run_copy(CC, A,   N, 16, 0, 8, L0, stream);
    run_copy(CC, OG1, N, 16, 8, 8, L0, stream);
    run_conv(CC, F(61), nullptr, B, N, 16, 8, 64, 64, 64, 0, W16, stream);
    run_bn(B, N, 8, 64, 64, 64, F(62), F(63), 1, STATS, PART, stream);
    run_conv(B, F(64), nullptr, A, N, 8, 8, 64, 64, 64, 0, W16, stream);
    run_bn(A, N, 8, 64, 64, 64, F(65), F(66), 1, STATS, PART, stream);

    // ---------------- post-filter + final conv ----------------
    run_conv(A, F(67), nullptr, B, N, 8, 8, 64, 64, 64, 0, W16, stream);
    run_bn(B, N, 8, 64, 64, 64, F(68), F(69), 0, STATS, PART, stream);
    run_conv(B, F(70), F(71), (float*)d_out, N, 8, 32, 64, 64, 64, 1, W16, stream);

    #undef F
}
